// DPASSMBlock_44951127720633
// MI455X (gfx1250) — matrix operand order repaired and checked by reading, not compiled, not run on hardware
//
#include <hip/hip_runtime.h>

// ---------------------------------------------------------------------------
// DPA+SSM block for gfx1250 (MI455X).
// All GEMM-class math on v_wmma_f32_16x16x32_bf16 (fp32 accumulate) with
// pre-converted bf16 operands; GEMM A-panels staged into LDS via the Tensor
// Data Mover (tensor_load_to_lds / s_wait_tensorcnt) when available.
// B=2, T=2048, D=1024, H=16, d_h=64, WINDOW=256, N_STATE=128.
// ---------------------------------------------------------------------------

typedef __attribute__((ext_vector_type(16))) __bf16 v16bf;
typedef __attribute__((ext_vector_type(8)))  float  v8f;
typedef __attribute__((ext_vector_type(4)))  unsigned int v4u;
typedef __attribute__((ext_vector_type(8)))  int    v8i;
typedef __attribute__((ext_vector_type(4)))  int    v4i;

typedef unsigned short u16;
typedef unsigned int   u32;

union Frag { u32 u[8]; v16bf v; };

static __device__ __forceinline__ u16 f2bf(float f) {
  __bf16 h = (__bf16)f;
  return __builtin_bit_cast(u16, h);
}
static __device__ __forceinline__ u32 pk2(float a, float b) {
  return (u32)f2bf(a) | ((u32)f2bf(b) << 16);
}

#if __has_builtin(__builtin_amdgcn_tensor_load_to_lds)
#define USE_TDM 1
#else
#define USE_TDM 0
#endif

// ---------------------------------------------------------------------------
// fp32 -> bf16 bulk conversion (weights / activations), 8 elems per thread.
// ---------------------------------------------------------------------------
__global__ void __launch_bounds__(256) cvt_bf16_kernel(const float* __restrict__ in,
                                                       u16* __restrict__ out, int n) {
  int i8 = (blockIdx.x * 256 + threadIdx.x) * 8;
  if (i8 < n) {
    float4 f0 = *reinterpret_cast<const float4*>(in + i8);
    float4 f1 = *reinterpret_cast<const float4*>(in + i8 + 4);
    uint4 o;
    o.x = pk2(f0.x, f0.y); o.y = pk2(f0.z, f0.w);
    o.z = pk2(f1.x, f1.y); o.w = pk2(f1.z, f1.w);
    *reinterpret_cast<uint4*>(out + i8) = o;
  }
}

// ---------------------------------------------------------------------------
// LayerNorm: one 256-thread block per row of d=1024. Output bf16 (GEMM feed).
// ---------------------------------------------------------------------------
__global__ void __launch_bounds__(256) ln_kernel(const float* __restrict__ x,
                                                 const float* __restrict__ g,
                                                 const float* __restrict__ b,
                                                 u16* __restrict__ out, int d) {
  __shared__ float red[256];
  const size_t row = blockIdx.x;
  const float* xr = x + row * (size_t)d;
  const int tid = threadIdx.x;

  float s = 0.f;
#pragma unroll
  for (int l = 0; l < 4; ++l) s += xr[tid + 256 * l];
  red[tid] = s; __syncthreads();
  for (int st = 128; st > 0; st >>= 1) {
    if (tid < st) red[tid] += red[tid + st];
    __syncthreads();
  }
  const float mu = red[0] / (float)d;
  __syncthreads();

  float vs = 0.f;
#pragma unroll
  for (int l = 0; l < 4; ++l) { float t = xr[tid + 256 * l] - mu; vs += t * t; }
  red[tid] = vs; __syncthreads();
  for (int st = 128; st > 0; st >>= 1) {
    if (tid < st) red[tid] += red[tid + st];
    __syncthreads();
  }
  const float rstd = rsqrtf(red[0] / (float)d + 1e-5f);

#pragma unroll
  for (int l = 0; l < 4; ++l) {
    int c = tid + 256 * l;
    out[row * (size_t)d + c] = f2bf((xr[c] - mu) * rstd * g[c] + b[c]);
  }
}

// ---------------------------------------------------------------------------
// Tiled GEMM, bf16 in / fp32 acc: C = act(A[M,K] @ W[K,N] + bias) (+ res).
// BM=128 BN=64 BK=32, 256 thr = 8 waves (4x2), wave = 32x32 (2x2 WMMA tiles).
// A-panel staged via TDM (wave 0) while other waves transpose the B-panel.
// ACT: 0=none 1=sigmoid 2=exact gelu. OBF: bf16 output (feeds another GEMM).
// ---------------------------------------------------------------------------
#define BM 128
#define BN 64
#define BK 32
#define LDT 40   // LDS row stride in halfs (multiple of 8 -> b128-aligned rows)

template <int ACT, bool RESID, bool OBF>
__global__ void __launch_bounds__(256) gemm_kernel(
    const u16* __restrict__ A, const u16* __restrict__ W,
    const float* __restrict__ bias, const float* __restrict__ res,
    void* __restrict__ Cv, int Mrows, int N, int K) {
  __shared__ u16 As[BM * LDT];
  __shared__ u16 Bs[BN * LDT];

  const int tid  = threadIdx.x;
  const int lane = tid & 31;
  const int wave = tid >> 5;
  const int wm   = wave >> 1;   // 0..3
  const int wn   = wave & 1;    // 0..1
  const int m0   = blockIdx.y * BM;
  const int n0   = blockIdx.x * BN;
  const int mlo  = lane & 15;
  const int hi   = lane >> 4;

  v8f acc[2][2];
#pragma unroll
  for (int mi = 0; mi < 2; ++mi)
#pragma unroll
    for (int ni = 0; ni < 2; ++ni)
#pragma unroll
      for (int r = 0; r < 8; ++r) acc[mi][ni][r] = 0.f;

#if USE_TDM
  const u32 lds_off = (u32)(uintptr_t)(&As[0]);
#endif

  for (int kk = 0; kk < K; kk += BK) {
#if USE_TDM
    // --- A tile: Tensor Data Mover, 128x32 bf16, LDS rows padded to 40 halfs
    if (wave == 0) {
      unsigned long long ga =
          (unsigned long long)(uintptr_t)(A + (size_t)m0 * K + kk);
      v4u g0 = {1u,                                   // count=1 (user D#)
                lds_off,                              // lds_addr
                (u32)(ga & 0xffffffffu),              // global_addr[31:0]
                (u32)((ga >> 32) & 0x1ffffffu) | (2u << 30)}; // addr[56:32] | type=2
      v8i g1 = {(int)((1u << 16) |                    // data_size = 2B
                      (1u << 20) |                    // pad_enable
                      (3u << 22) |                    // pad_interval = 16 DWORDs (64B row)
                      (3u << 25)),                    // pad_amount   = 4 DWORDs (8 halfs)
                (int)((u32)K << 16),                  // tensor_dim0[15:0] @ bit48
                (int)(((u32)K >> 16) | ((u32)Mrows << 16)),  // dim0 hi | dim1 lo
                (int)(((u32)Mrows >> 16) | (32u << 16)),     // dim1 hi | tile_dim0=32
                128,                                  // tile_dim1=128, tile_dim2=0
                (int)(u32)K,                          // tensor_dim0_stride[31:0]
                0, 0};
      v4i z4 = {0, 0, 0, 0};
#if defined(__clang_major__) && __clang_major__ >= 23
      v8i z8 = {0, 0, 0, 0, 0, 0, 0, 0};
      __builtin_amdgcn_tensor_load_to_lds(g0, g1, z4, z4, z8, 0);
#else
      __builtin_amdgcn_tensor_load_to_lds(g0, g1, z4, z4, 0);
#endif
    }
#else
    // --- A tile: manual b128 copy (bf16, no conversion)
#pragma unroll
    for (int l = 0; l < 2; ++l) {
      int s = tid + 256 * l;                  // 0..511 8-elem chunks
      int row = s >> 2, c8 = (s & 3) << 3;
      *reinterpret_cast<uint4*>(&As[row * LDT + c8]) =
          *reinterpret_cast<const uint4*>(A + (size_t)(m0 + row) * K + kk + c8);
    }
#endif

    // --- B tile: transpose 2x2 bf16 blocks into LDS [n][k]
#pragma unroll
    for (int l = 0; l < 2; ++l) {
      int u = tid + 256 * l;                  // 0..511 2x2 blocks
      int kp = (u >> 5) << 1;                 // 0..30 even
      int np = (u & 31) << 1;                 // 0..62 even
      u32 a = *reinterpret_cast<const u32*>(W + (size_t)(kk + kp) * N + n0 + np);
      u32 b = *reinterpret_cast<const u32*>(W + (size_t)(kk + kp + 1) * N + n0 + np);
      u32 p0 = (a & 0xffffu) | (b << 16);
      u32 p1 = (a >> 16) | (b & 0xffff0000u);
      *reinterpret_cast<u32*>(&Bs[(np + 0) * LDT + kp]) = p0;
      *reinterpret_cast<u32*>(&Bs[(np + 1) * LDT + kp]) = p1;
    }

#if USE_TDM
    if (wave == 0) __builtin_amdgcn_s_wait_tensorcnt(0);
#endif
    __syncthreads();

    Frag af[2], bf[2];
#pragma unroll
    for (int mi = 0; mi < 2; ++mi) {
      int row = wm * 32 + mi * 16 + mlo;
#pragma unroll
      for (int i = 0; i < 8; ++i) {
        int k = ((i >> 2) << 4) + (hi << 3) + ((i & 3) << 1);   // A 16x32 layout
        af[mi].u[i] = *reinterpret_cast<const u32*>(&As[row * LDT + k]);
      }
    }
#pragma unroll
    for (int ni = 0; ni < 2; ++ni) {
      int col = wn * 32 + ni * 16 + mlo;
#pragma unroll
      for (int i = 0; i < 8; ++i) {
        int k = ((i >> 2) << 4) + (hi << 3) + ((i & 3) << 1);   // B 32x16 layout
        bf[ni].u[i] = *reinterpret_cast<const u32*>(&Bs[col * LDT + k]);
      }
    }
#pragma unroll
    for (int mi = 0; mi < 2; ++mi)
#pragma unroll
      for (int ni = 0; ni < 2; ++ni)
        acc[mi][ni] = __builtin_amdgcn_wmma_f32_16x16x32_bf16(
            false, af[mi].v, false, bf[ni].v, (short)0, acc[mi][ni], false, false);
    __syncthreads();
  }

#pragma unroll
  for (int mi = 0; mi < 2; ++mi)
#pragma unroll
    for (int ni = 0; ni < 2; ++ni) {
      int n = n0 + wn * 32 + ni * 16 + mlo;
      float bv = bias ? bias[n] : 0.f;
#pragma unroll
      for (int r = 0; r < 8; ++r) {
        int m = m0 + wm * 32 + mi * 16 + r + (hi << 3);
        float v = acc[mi][ni][r] + bv;
        if (ACT == 1) v = 1.f / (1.f + expf(-v));
        else if (ACT == 2) v = 0.5f * v * (1.f + erff(v * 0.70710678118f));
        if (RESID) v += res[(size_t)m * N + n];
        if (OBF) ((u16*)Cv)[(size_t)m * N + n] = f2bf(v);
        else     ((float*)Cv)[(size_t)m * N + n] = v;
      }
    }
}

// ---------------------------------------------------------------------------
// Windowed causal flash attention (bf16 Q/K/V, bf16 out). One wave per
// 16-query tile, 32-key steps; online softmax in registers (width-16 shfl);
// P re-layout (C->A frag) through per-wave LDS.
// ---------------------------------------------------------------------------
#define DH 64
#define WIN 256

__global__ void __launch_bounds__(128) attn_kernel(
    const u16* __restrict__ Qb, const u16* __restrict__ Kb,
    const u16* __restrict__ Vb, u16* __restrict__ Ob, int T, int H) {
  __shared__ u16 Pbuf[4][16 * 36];
  const int lane = threadIdx.x & 31;
  const int wave = threadIdx.x >> 5;
  const int mlo  = lane & 15;
  const int hi   = lane >> 4;
  const int q0   = (blockIdx.x * 4 + wave) * 16;
  const int b    = blockIdx.y >> 4;   // H == 16
  const int h    = blockIdx.y & 15;
  const int d    = H * DH;
  const size_t rowbase = (size_t)b * T;

  // Q fragments: direct packed-bf16 loads (16x64 A-matrix = two 16x32 frags)
  Frag qf[2];
#pragma unroll
  for (int j = 0; j < 2; ++j)
#pragma unroll
    for (int i = 0; i < 8; ++i) {
      int c = j * 32 + ((i >> 2) << 4) + (hi << 3) + ((i & 3) << 1);
      qf[j].u[i] = *reinterpret_cast<const u32*>(
          Qb + (rowbase + q0 + mlo) * d + h * DH + c);
    }

  v8f o[4];
#pragma unroll
  for (int nt = 0; nt < 4; ++nt)
#pragma unroll
    for (int r = 0; r < 8; ++r) o[nt][r] = 0.f;
  float mrow[8], lrow[8];
#pragma unroll
  for (int r = 0; r < 8; ++r) { mrow[r] = -1e30f; lrow[r] = 0.f; }

  int kb_min = q0 - (WIN - 1); if (kb_min < 0) kb_min = 0;
  const int kb_start = kb_min & ~31;

  for (int kb = kb_start; kb <= q0 + 15; kb += 32) {
    // S = Q @ K^T for 32 keys (two 16-key WMMA n-tiles)
    v8f s[2];
#pragma unroll
    for (int t = 0; t < 2; ++t) {
#pragma unroll
      for (int r = 0; r < 8; ++r) s[t][r] = 0.f;
#pragma unroll
      for (int j = 0; j < 2; ++j) {
        Frag kf;
#pragma unroll
        for (int i = 0; i < 8; ++i) {
          int c = j * 32 + ((i >> 2) << 4) + (hi << 3) + ((i & 3) << 1);
          int key = kb + t * 16 + mlo; if (key > T - 1) key = T - 1;
          kf.u[i] = *reinterpret_cast<const u32*>(
              Kb + (rowbase + key) * d + h * DH + c);
        }
        s[t] = __builtin_amdgcn_wmma_f32_16x16x32_bf16(
            false, qf[j].v, false, kf.v, (short)0, s[t], false, false);
      }
    }

    // scale + window mask, per-row running max
    float tmax[8];
#pragma unroll
    for (int r = 0; r < 8; ++r) {
      const int iq = q0 + r + (hi << 3);
      float vmax = -1e30f;
#pragma unroll
      for (int t = 0; t < 2; ++t) {
        int j = kb + t * 16 + mlo;
        float sv = s[t][r] * 0.125f;                       // 1/sqrt(64)
        bool ok = (j <= iq) && (iq - j < WIN);
        sv = ok ? sv : -1e30f;
        s[t][r] = sv;
        vmax = fmaxf(vmax, sv);
      }
#pragma unroll
      for (int off = 1; off < 16; off <<= 1)
        vmax = fmaxf(vmax, __shfl_xor(vmax, off, 16));
      tmax[r] = vmax;
    }

    float alpha[8];
#pragma unroll
    for (int r = 0; r < 8; ++r) {
      float mn = fmaxf(mrow[r], tmax[r]);
      alpha[r] = expf(mrow[r] - mn);
      mrow[r] = mn;
    }

    // P = exp(S - m), row sums, stage bf16 P to LDS for A-frag re-layout
    float psum[8];
#pragma unroll
    for (int r = 0; r < 8; ++r) {
      const int iq = q0 + r + (hi << 3);
      float ps = 0.f;
#pragma unroll
      for (int t = 0; t < 2; ++t) {
        int j = kb + t * 16 + mlo;
        bool ok = (j <= iq) && (iq - j < WIN);
        float pe = ok ? expf(s[t][r] - mrow[r]) : 0.f;
        ps += pe;
        Pbuf[wave][(r + (hi << 3)) * 36 + t * 16 + mlo] = f2bf(pe);
      }
#pragma unroll
      for (int off = 1; off < 16; off <<= 1) ps += __shfl_xor(ps, off, 16);
      psum[r] = ps;
    }
#pragma unroll
    for (int r = 0; r < 8; ++r) lrow[r] = lrow[r] * alpha[r] + psum[r];
#pragma unroll
    for (int nt = 0; nt < 4; ++nt)
#pragma unroll
      for (int r = 0; r < 8; ++r) o[nt][r] *= alpha[r];

    asm volatile("s_wait_dscnt 0" ::: "memory");   // LDS RAW, same wave

    Frag pf;
#pragma unroll
    for (int i = 0; i < 8; ++i) {
      int k = ((i >> 2) << 4) + (hi << 3) + ((i & 3) << 1);
      pf.u[i] = *reinterpret_cast<const u32*>(&Pbuf[wave][mlo * 36 + k]);
    }

    // O += P(16x32) @ V(32x64); V pairs span two rows -> d16 lo/hi pack
#pragma unroll
    for (int nt = 0; nt < 4; ++nt) {
      Frag vf;
#pragma unroll
      for (int i = 0; i < 8; ++i) {
        int k0 = kb + ((i >> 2) << 4) + (hi << 3) + ((i & 3) << 1);
        int ra = (k0     > T - 1) ? T - 1 : k0;
        int rb = (k0 + 1 > T - 1) ? T - 1 : k0 + 1;
        u32 va  = Vb[(rowbase + ra) * d + h * DH + nt * 16 + mlo];
        u32 vb2 = Vb[(rowbase + rb) * d + h * DH + nt * 16 + mlo];
        vf.u[i] = va | (vb2 << 16);
      }
      o[nt] = __builtin_amdgcn_wmma_f32_16x16x32_bf16(
          false, pf.v, false, vf.v, (short)0, o[nt], false, false);
    }
  }

#pragma unroll
  for (int r = 0; r < 8; ++r) {
    float inv = 1.0f / lrow[r];
    int m = q0 + r + (hi << 3);
#pragma unroll
    for (int nt = 0; nt < 4; ++nt)
      Ob[(rowbase + m) * d + h * DH + nt * 16 + mlo] = f2bf(o[nt][r] * inv);
  }
}

// ---------------------------------------------------------------------------
// Diagonal SSM scan: 256 independent recurrences (b, n), length T.
// states written as bf16 (feeds Cw GEMM); final state f32 to d_out tail.
// ---------------------------------------------------------------------------
__global__ void __launch_bounds__(256) scan_kernel(
    const float* __restrict__ u, const float* __restrict__ A,
    const float* __restrict__ s0, u16* __restrict__ states,
    float* __restrict__ out_state, int T) {
  const int tid = threadIdx.x;
  const int b = tid >> 7, n = tid & 127;
  const float a = A[n];
  float s = s0[b * 128 + n];
  const float* up = u + (size_t)b * T * 128 + n;
  u16* sp = states + (size_t)b * T * 128 + n;
  for (int t = 0; t < T; ++t) {
    s = fmaf(a, s, up[(size_t)t * 128]);
    sp[(size_t)t * 128] = f2bf(s);
  }
  out_state[b * 128 + n] = s;
}

// ---------------------------------------------------------------------------
// Gated fusion: x + g*attn + (1-g)*ssm  (all f32)
// ---------------------------------------------------------------------------
__global__ void __launch_bounds__(256) fusion_kernel(
    const float* __restrict__ x, const float* __restrict__ g,
    const float* __restrict__ at, const float* __restrict__ sm,
    float* __restrict__ out, size_t n) {
  size_t i = (size_t)blockIdx.x * 256 + threadIdx.x;
  if (i < n) {
    float gv = g[i];
    out[i] = x[i] + gv * at[i] + (1.f - gv) * sm[i];
  }
}

// ---------------------------------------------------------------------------
// Host orchestration
// ---------------------------------------------------------------------------
extern "C" void kernel_launch(void* const* d_in, const int* in_sizes, int n_in,
                              void* d_out, int out_size, void* d_ws, size_t ws_size,
                              hipStream_t stream) {
  (void)in_sizes; (void)n_in; (void)out_size; (void)ws_size;
  constexpr int Bb = 2, T = 2048, D = 1024, H = 16, NS = 128, HID = 4096;
  constexpr int M = Bb * T;                 // 4096 rows
  const size_t S = (size_t)M * D;           // elements per [M,D]

  const float* x    = (const float*)d_in[0];
  const float* st0  = (const float*)d_in[1];
  const float* l1g  = (const float*)d_in[2];
  const float* l1b  = (const float*)d_in[3];
  const float* wq   = (const float*)d_in[4];
  const float* bq   = (const float*)d_in[5];
  const float* wk   = (const float*)d_in[6];
  const float* bk   = (const float*)d_in[7];
  const float* wv   = (const float*)d_in[8];
  const float* bv   = (const float*)d_in[9];
  const float* wo   = (const float*)d_in[10];
  const float* bo   = (const float*)d_in[11];
  const float* wg   = (const float*)d_in[12];
  const float* bg   = (const float*)d_in[13];
  const float* Avec = (const float*)d_in[14];
  const float* Bw   = (const float*)d_in[15];
  const float* Cw   = (const float*)d_in[16];
  const float* l2g  = (const float*)d_in[17];
  const float* l2b  = (const float*)d_in[18];
  const float* w1   = (const float*)d_in[19];
  const float* b1   = (const float*)d_in[20];
  const float* w2   = (const float*)d_in[21];
  const float* b2   = (const float*)d_in[22];

  // Workspace carve-out (~134 MB). 256B-aligned slabs.
  char* wsb = (char*)d_ws;
  size_t off = 0;
  auto alloc = [&](size_t bytes) -> void* {
    void* p = wsb + off;
    off += (bytes + 255) & ~(size_t)255;
    return p;
  };
  u16*  xnb  = (u16*)alloc(S * 2);              // LN1/LN2 output (bf16)
  u16*  qb   = (u16*)alloc(S * 2);              // Q (bf16)
  u16*  kb   = (u16*)alloc(S * 2);              // K (bf16)
  u16*  vb   = (u16*)alloc(S * 2);              // V (bf16)
  u16*  aob  = (u16*)alloc(S * 2);              // attn pre-wo (bf16)
  float* gbuf = (float*)alloc(S * 4);           // sigmoid gate
  float* attn = (float*)alloc(S * 4);           // attn post-wo
  float* ssmb = (float*)alloc(S * 4);           // ssm_out
  float* xmid = (float*)alloc(S * 4);           // fused residual
  float* ub   = (float*)alloc((size_t)M * NS * 4);
  u16*  stb  = (u16*)alloc((size_t)M * NS * 2); // scan states (bf16)
  u16*  wqb  = (u16*)alloc((size_t)D * D * 2);
  u16*  wkb  = (u16*)alloc((size_t)D * D * 2);
  u16*  wvb  = (u16*)alloc((size_t)D * D * 2);
  u16*  wgb  = (u16*)alloc((size_t)D * D * 2);
  u16*  wob  = (u16*)alloc((size_t)D * D * 2);
  u16*  Bwb  = (u16*)alloc((size_t)D * NS * 2);
  u16*  Cwb  = (u16*)alloc((size_t)NS * D * 2);
  u16*  w1b  = (u16*)alloc((size_t)D * HID * 2);
  u16*  w2b  = (u16*)alloc((size_t)HID * D * 2);
  u16*  hbuf = qb;   // MLP hidden [M,HID] bf16 (32MB) aliases q/k/v/ao slabs
  u16*  xn2b = xnb;

  float* out_x  = (float*)d_out;
  float* out_st = out_x + S;

  const dim3 blk(256);
  const dim3 gD(D / BN, M / BM);      // N=1024
  const dim3 gNS(NS / BN, M / BM);    // N=128
  const dim3 gH(HID / BN, M / BM);    // N=4096
  auto cvt = [&](const float* src, u16* dst, int n) {
    cvt_bf16_kernel<<<n / 2048, blk, 0, stream>>>(src, dst, n);
  };

  // 0) weight conversions (bf16 copies stay L2-resident: ~26 MB total)
  cvt(wq, wqb, D * D);   cvt(wk, wkb, D * D);   cvt(wv, wvb, D * D);
  cvt(wg, wgb, D * D);   cvt(wo, wob, D * D);
  cvt(Bw, Bwb, D * NS);  cvt(Cw, Cwb, NS * D);
  cvt(w1, w1b, D * HID); cvt(w2, w2b, HID * D);

  // 1) LN1 -> bf16
  ln_kernel<<<M, blk, 0, stream>>>(x, l1g, l1b, xnb, D);
  // 2) Q/K/V (bf16 out) and gate (f32 sigmoid)
  gemm_kernel<0, false, true ><<<gD, blk, 0, stream>>>(xnb, wqb, bq, nullptr, qb, M, D, D);
  gemm_kernel<0, false, true ><<<gD, blk, 0, stream>>>(xnb, wkb, bk, nullptr, kb, M, D, D);
  gemm_kernel<0, false, true ><<<gD, blk, 0, stream>>>(xnb, wvb, bv, nullptr, vb, M, D, D);
  gemm_kernel<1, false, false><<<gD, blk, 0, stream>>>(xnb, wgb, bg, nullptr, gbuf, M, D, D);
  // 3) SSM input projection u = xn @ Bw
  gemm_kernel<0, false, false><<<gNS, blk, 0, stream>>>(xnb, Bwb, nullptr, nullptr, ub, M, NS, D);
  // 4) windowed attention (bf16 in/out)
  attn_kernel<<<dim3(T / 64, Bb * H), dim3(128), 0, stream>>>(qb, kb, vb, aob, T, H);
  // 5) output projection
  gemm_kernel<0, false, false><<<gD, blk, 0, stream>>>(aob, wob, bo, nullptr, attn, M, D, D);
  // 6) SSM scan (writes new_state into d_out tail)
  scan_kernel<<<1, blk, 0, stream>>>(ub, Avec, st0, stb, out_st, T);
  // 7) ssm_out = states @ Cw (K=128)
  gemm_kernel<0, false, false><<<gD, blk, 0, stream>>>(stb, Cwb, nullptr, nullptr, ssmb, M, D, NS);
  // 8) gated fusion + residual
  fusion_kernel<<<(unsigned)((S + 255) / 256), blk, 0, stream>>>(x, gbuf, attn, ssmb, xmid, S);
  // 9) LN2 -> bf16
  ln_kernel<<<M, blk, 0, stream>>>(xmid, l2g, l2b, xn2b, D);
  // 10) MLP up, exact GELU fused, bf16 out (feeds final GEMM)
  gemm_kernel<2, false, true ><<<gH, blk, 0, stream>>>(xn2b, w1b, b1, nullptr, hbuf, M, HID, D);
  // 11) MLP down + residual, writes final x output (f32) to d_out
  gemm_kernel<0, true, false><<<gD, blk, 0, stream>>>(hbuf, w2b, b2, xmid, out_x, M, D, HID);
}
